// GRU_67912022884942
// MI455X (gfx1250) — matrix operand order repaired and checked by reading, not compiled, not run on hardware
//
#include <hip/hip_runtime.h>
#include <hip/hip_bf16.h>
#include <cstddef>
#include <cstdint>

// ---------------- problem constants ----------------
#define S_  512
#define B_  32
#define I_  256
#define H_  512
#define V_  10000
#define H3  1536        // 3*H
#define C2H 1024        // 2*H
#define M_  (S_ * B_)   // 16384 rows when time*batch flattened

typedef __attribute__((ext_vector_type(16))) __bf16 v16bf;
typedef __attribute__((ext_vector_type(8)))  float  v8f;

// ---------------- helpers ----------------
__device__ __forceinline__ unsigned short f2bf(float f) {
  union { float f; unsigned u; } c; c.f = f;
  unsigned u = c.u;
  unsigned r = u + 0x7FFFu + ((u >> 16) & 1u);   // round-to-nearest-even
  return (unsigned short)(r >> 16);
}

union TileU { uint4 q[2]; v16bf v; };

// A tile (16 x 32, 16-bit): lane m<16 holds row m K=[0..7]+[16..23];
// lane m+16 holds row m K=[8..15]+[24..31]  (ISA 7.12.2 16-bit A layout)
__device__ __forceinline__ v16bf load_tile_A(const unsigned short* base, int ld) {
  int lane = threadIdx.x & 31;
  int row  = lane & 15;
  int koff = (lane >> 4) * 8;
  const unsigned short* p = base + (size_t)row * ld + koff;
  TileU t;
  t.q[0] = *(const uint4*)(p);
  t.q[1] = *(const uint4*)(p + 16);
  return t.v;
}

// B tile (32 x 16, 16-bit) from row-major weight W[N][ld] (so B = W^T):
// lane n<16 holds column n K=[0..7]+[16..23]; lane n+16 holds column n K=[8..15]+[24..31]
__device__ __forceinline__ v16bf load_tile_B(const unsigned short* w, int ld) {
  int lane = threadIdx.x & 31;
  int n    = lane & 15;
  int krun = (lane >> 4) * 8;
  const unsigned short* p = w + (size_t)n * ld + krun;
  TileU t;
  t.q[0] = *(const uint4*)(p);
  t.q[1] = *(const uint4*)(p + 16);
  return t.v;
}

__device__ __forceinline__ v8f wmma_bf16(v16bf a, v16bf b, v8f c) {
  return __builtin_amdgcn_wmma_f32_16x16x32_bf16(false, a, false, b,
                                                 (short)0, c, false, false);
}

__device__ __forceinline__ float sigmoidf_(float x) {
  return 1.0f / (1.0f + __expf(-x));
}
__device__ __forceinline__ float tanhf_(float x) {
  float e = __expf(2.0f * x);
  return (e - 1.0f) / (e + 1.0f);
}

// ---------------- kernels ----------------
__global__ void cast_f32_bf16(const float* __restrict__ src,
                              unsigned short* __restrict__ dst, int n) {
  int i = blockIdx.x * blockDim.x + threadIdx.x;
  if (i < n) dst[i] = f2bf(src[i]);
}

// C[M,N] f32 = A[M,K] bf16 @ W[N,K]^T bf16 + bias[N]
// block = 256 threads = 8 waves arranged 2(M) x 4(N).
// wave tile: 32 rows x 64 cols (2 M-tiles x 4 N-tiles, 8 accumulators).
// block tile: 64 x 256.  grid = (ceil(N/256), M/64).
// Requires K%32==0, N%16==0, M%64==0. N tail handled with uniform guards.
__global__ __launch_bounds__(256)
void gemm_bf16_wmma(const unsigned short* __restrict__ A,
                    const unsigned short* __restrict__ W,
                    const float* __restrict__ bias,
                    float* __restrict__ C, int K, int N) {
  int wave = threadIdx.x >> 5;
  int lane = threadIdx.x & 31;
  int mw = wave >> 2;                      // 0..1
  int nw = wave & 3;                       // 0..3
  int row0 = blockIdx.y * 64 + mw * 32;
  int n0   = blockIdx.x * 256 + nw * 64;   // wave's first column
  const unsigned short* Arow = A + (size_t)row0 * K;

  // clamp weight strips in-bounds; validity is wave-uniform (EXEC stays all-1)
  const unsigned short* Wn[4];
  bool valid[4];
#pragma unroll
  for (int nt = 0; nt < 4; ++nt) {
    int nc = n0 + nt * 16;
    valid[nt] = (nc < N);
    Wn[nt] = W + (size_t)(valid[nt] ? nc : (N - 16)) * K;
  }

  v8f acc[2][4];
#pragma unroll
  for (int mt = 0; mt < 2; ++mt)
#pragma unroll
    for (int nt = 0; nt < 4; ++nt) acc[mt][nt] = (v8f){};

  // ---- software-pipelined k loop: load k+32 while WMMAs consume k ----
  v16bf a0 = load_tile_A(Arow, K);
  v16bf a1 = load_tile_A(Arow + 16 * (size_t)K, K);
  v16bf b0 = load_tile_B(Wn[0], K);
  v16bf b1 = load_tile_B(Wn[1], K);
  v16bf b2 = load_tile_B(Wn[2], K);
  v16bf b3 = load_tile_B(Wn[3], K);

  for (int k0 = 32; k0 < K; k0 += 32) {
    if (k0 + 32 < K)
      __builtin_prefetch(Wn[0] + (size_t)(lane & 15) * K + k0 + 32, 0, 1);
    v16bf a0n = load_tile_A(Arow + k0, K);
    v16bf a1n = load_tile_A(Arow + 16 * (size_t)K + k0, K);
    v16bf b0n = load_tile_B(Wn[0] + k0, K);
    v16bf b1n = load_tile_B(Wn[1] + k0, K);
    v16bf b2n = load_tile_B(Wn[2] + k0, K);
    v16bf b3n = load_tile_B(Wn[3] + k0, K);
    acc[0][0] = wmma_bf16(a0, b0, acc[0][0]);
    acc[1][0] = wmma_bf16(a1, b0, acc[1][0]);
    acc[0][1] = wmma_bf16(a0, b1, acc[0][1]);
    acc[1][1] = wmma_bf16(a1, b1, acc[1][1]);
    acc[0][2] = wmma_bf16(a0, b2, acc[0][2]);
    acc[1][2] = wmma_bf16(a1, b2, acc[1][2]);
    acc[0][3] = wmma_bf16(a0, b3, acc[0][3]);
    acc[1][3] = wmma_bf16(a1, b3, acc[1][3]);
    a0 = a0n; a1 = a1n; b0 = b0n; b1 = b1n; b2 = b2n; b3 = b3n;
  }
  acc[0][0] = wmma_bf16(a0, b0, acc[0][0]);
  acc[1][0] = wmma_bf16(a1, b0, acc[1][0]);
  acc[0][1] = wmma_bf16(a0, b1, acc[0][1]);
  acc[1][1] = wmma_bf16(a1, b1, acc[1][1]);
  acc[0][2] = wmma_bf16(a0, b2, acc[0][2]);
  acc[1][2] = wmma_bf16(a1, b2, acc[1][2]);
  acc[0][3] = wmma_bf16(a0, b3, acc[0][3]);
  acc[1][3] = wmma_bf16(a1, b3, acc[1][3]);

  int hi = lane >> 4;
  int lane15 = lane & 15;
#pragma unroll
  for (int nt = 0; nt < 4; ++nt) {
    if (valid[nt]) {
      int n = n0 + nt * 16 + lane15;
      float bv = bias ? bias[n] : 0.0f;
      float* Crow = C + (size_t)(row0 + hi * 8) * N + n;
#pragma unroll
      for (int r = 0; r < 8; ++r) {
        Crow[(size_t)r * N]        = acc[0][nt][r] + bv;  // rows row0+hi*8+r
        Crow[(size_t)(16 + r) * N] = acc[1][nt][r] + bv;  // rows +16
      }
    }
  }
}

// One GRU direction scan. grid.x = 2 (dir), block = 1024 threads (32 waves).
// Hidden state lives in LDS; per step: gh = h @ Whh^T via WMMA (each wave owns
// matching 16-col strips of r/z/n -> gates fused in registers), then h update.
__global__ __launch_bounds__(1024)
void gru_scan(const float* __restrict__ xp,            // [2][M][3H] f32 (this layer)
              const unsigned short* __restrict__ Whh,  // [2][3H][H] bf16
              const float* __restrict__ bhh,           // [2][3H]
              const float* __restrict__ h0,            // [2][B][H] (this layer)
              unsigned short* __restrict__ ycat,       // [S][B][2H] bf16
              float* __restrict__ hfin)                // [2][B][H] f32 (this layer)
{
  int d = blockIdx.x;                    // 0 = forward, 1 = backward
  xp   += (size_t)d * M_ * H3;
  Whh  += (size_t)d * H3 * H_;
  bhh  += (size_t)d * H3;
  h0   += (size_t)d * B_ * H_;
  hfin += (size_t)d * B_ * H_;
  int  colOff = d * H_;
  bool rev    = (d == 1);

  extern __shared__ char smem[];
  float*          hS = (float*)smem;                                  // 64 KB
  unsigned short* hB = (unsigned short*)(smem + B_ * H_ * sizeof(float)); // 32 KB

  int tid = threadIdx.x;
  for (int i = tid; i < B_ * H_; i += blockDim.x) {
    float v = h0[i];
    hS[i] = v;
    hB[i] = f2bf(v);
  }
  __syncthreads();

  int wave   = tid >> 5;
  int lane   = tid & 31;
  int lane15 = lane & 15;
  int hi     = lane >> 4;
  int jcol   = wave * 16 + lane15;       // hidden index 0..511 owned by this lane

  float bR = bhh[jcol];
  float bZ = bhh[H_ + jcol];
  float bN = bhh[2 * H_ + jcol];

  const unsigned short* WhR = Whh + (size_t)(0 * H_ + wave * 16) * H_;
  const unsigned short* WhZ = Whh + (size_t)(1 * H_ + wave * 16) * H_;
  const unsigned short* WhN = Whh + (size_t)(2 * H_ + wave * 16) * H_;

  for (int t = 0; t < S_; ++t) {
    int s = rev ? (S_ - 1 - t) : t;
    v8f ar0 = {}, ar1 = {}, az0 = {}, az1 = {}, an0 = {}, an1 = {};

    // ---- software-pipelined K loop over H_ (16 steps of 32) ----
    v16bf a0 = load_tile_A(hB, H_);                  // batch rows 0..15
    v16bf a1 = load_tile_A(hB + 16 * H_, H_);        // batch rows 16..31
    v16bf tR = load_tile_B(WhR, H_);
    v16bf tZ = load_tile_B(WhZ, H_);
    v16bf tN = load_tile_B(WhN, H_);
    for (int k0 = 32; k0 < H_; k0 += 32) {
      v16bf a0n = load_tile_A(hB + k0, H_);
      v16bf a1n = load_tile_A(hB + 16 * H_ + k0, H_);
      v16bf tRn = load_tile_B(WhR + k0, H_);
      v16bf tZn = load_tile_B(WhZ + k0, H_);
      v16bf tNn = load_tile_B(WhN + k0, H_);
      ar0 = wmma_bf16(a0, tR, ar0);  ar1 = wmma_bf16(a1, tR, ar1);
      az0 = wmma_bf16(a0, tZ, az0);  az1 = wmma_bf16(a1, tZ, az1);
      an0 = wmma_bf16(a0, tN, an0);  an1 = wmma_bf16(a1, tN, an1);
      a0 = a0n; a1 = a1n; tR = tRn; tZ = tZn; tN = tNn;
    }
    ar0 = wmma_bf16(a0, tR, ar0);  ar1 = wmma_bf16(a1, tR, ar1);
    az0 = wmma_bf16(a0, tZ, az0);  az1 = wmma_bf16(a1, tZ, az1);
    an0 = wmma_bf16(a0, tN, an0);  an1 = wmma_bf16(a1, tN, an1);

    __syncthreads();   // every wave finished reading hB before owners rewrite it

    const float* xps = xp + (size_t)(s * B_) * H3;
#pragma unroll
    for (int mt = 0; mt < 2; ++mt) {
      v8f vr = mt ? ar1 : ar0;
      v8f vz = mt ? az1 : az0;
      v8f vn = mt ? an1 : an0;
#pragma unroll
      for (int r = 0; r < 8; ++r) {
        int b = mt * 16 + hi * 8 + r;                 // batch index
        const float* xpb = xps + (size_t)b * H3;
        float rr = sigmoidf_(xpb[jcol]          + vr[r] + bR);
        float zz = sigmoidf_(xpb[H_ + jcol]     + vz[r] + bZ);
        float nn = tanhf_(xpb[2 * H_ + jcol] + rr * (vn[r] + bN));
        float hold = hS[b * H_ + jcol];
        float hnew = (1.0f - zz) * nn + zz * hold;
        hS[b * H_ + jcol] = hnew;                     // exclusive owner
        hB[b * H_ + jcol] = f2bf(hnew);
        ycat[((size_t)(s * B_ + b)) * C2H + colOff + jcol] = f2bf(hnew);
      }
    }
    __syncthreads();   // h update visible before next step's WMMA reads
  }
  for (int i = tid; i < B_ * H_; i += blockDim.x) hfin[i] = hS[i];
}

// In-place softmax over V per (s,b) row. grid = M_, block = 256.
__global__ __launch_bounds__(256)
void softmax_rows(float* __restrict__ x) {
  __shared__ float red[256];
  float* p = x + (size_t)blockIdx.x * V_;
  int tid = threadIdx.x;
  float mx = -3.402823466e38f;
  for (int i = tid; i < V_; i += 256) mx = fmaxf(mx, p[i]);
  red[tid] = mx; __syncthreads();
  for (int s = 128; s > 0; s >>= 1) {
    if (tid < s) red[tid] = fmaxf(red[tid], red[tid + s]);
    __syncthreads();
  }
  mx = red[0]; __syncthreads();
  float sum = 0.0f;
  for (int i = tid; i < V_; i += 256) {
    float e = __expf(p[i] - mx);
    p[i] = e;
    sum += e;
  }
  red[tid] = sum; __syncthreads();
  for (int s = 128; s > 0; s >>= 1) {
    if (tid < s) red[tid] += red[tid + s];
    __syncthreads();
  }
  float inv = 1.0f / red[0];
  for (int i = tid; i < V_; i += 256) p[i] *= inv;
}

// ---------------- launcher ----------------
extern "C" void kernel_launch(void* const* d_in, const int* in_sizes, int n_in,
                              void* d_out, int out_size, void* d_ws, size_t ws_size,
                              hipStream_t stream) {
  (void)in_sizes; (void)n_in; (void)out_size; (void)ws_size;
  const float* feature = (const float*)d_in[0];
  const float* h       = (const float*)d_in[1];
  const float* Wih0    = (const float*)d_in[2];
  const float* Whh0    = (const float*)d_in[3];
  const float* bih0    = (const float*)d_in[4];
  const float* bhh0    = (const float*)d_in[5];
  const float* Wih1    = (const float*)d_in[6];
  const float* Whh1    = (const float*)d_in[7];
  const float* bih1    = (const float*)d_in[8];
  const float* bhh1    = (const float*)d_in[9];
  const float* Wlin    = (const float*)d_in[10];
  const float* blin    = (const float*)d_in[11];
  float* out = (float*)d_out;                       // [S][B][V] then [2L][B][H]
  float* hnext = out + (size_t)S_ * B_ * V_;

  // workspace carve-up (all chunks are multiples of 256 bytes)
  char* ws = (char*)d_ws;
  unsigned short* featB = (unsigned short*)ws;  ws += (size_t)M_ * I_ * 2;          // 8 MB
  unsigned short* wih0B = (unsigned short*)ws;  ws += (size_t)2 * H3 * I_ * 2;      // 1.5 MB
  unsigned short* whh0B = (unsigned short*)ws;  ws += (size_t)2 * H3 * H_ * 2;      // 3 MB
  unsigned short* wih1B = (unsigned short*)ws;  ws += (size_t)2 * H3 * C2H * 2;     // 6 MB
  unsigned short* whh1B = (unsigned short*)ws;  ws += (size_t)2 * H3 * H_ * 2;      // 3 MB
  unsigned short* wlinB = (unsigned short*)ws;  ws += (size_t)V_ * C2H * 2;         // 20.5 MB
  float*          xpbuf = (float*)ws;           ws += (size_t)2 * M_ * H3 * 4;      // 201 MB (reused)
  unsigned short* y0cat = (unsigned short*)ws;  ws += (size_t)M_ * C2H * 2;         // 33.5 MB
  unsigned short* y1cat = (unsigned short*)ws;  ws += (size_t)M_ * C2H * 2;         // 33.5 MB

  auto cast = [&](const float* s, unsigned short* d, int n) {
    cast_f32_bf16<<<(n + 255) / 256, 256, 0, stream>>>(s, d, n);
  };
  cast(feature, featB, M_ * I_);
  cast(Wih0, wih0B, 2 * H3 * I_);
  cast(Whh0, whh0B, 2 * H3 * H_);
  cast(Wih1, wih1B, 2 * H3 * C2H);
  cast(Whh1, whh1B, 2 * H3 * H_);
  cast(Wlin, wlinB, V_ * C2H);

  const size_t smemScan = (size_t)B_ * H_ * (sizeof(float) + sizeof(unsigned short)); // 96 KB

  // layer 0: batched input projection (both dirs), then scans
  for (int d = 0; d < 2; ++d) {
    gemm_bf16_wmma<<<dim3((H3 + 255) / 256, M_ / 64), 256, 0, stream>>>(
        featB, wih0B + (size_t)d * H3 * I_, bih0 + (size_t)d * H3,
        xpbuf + (size_t)d * M_ * H3, I_, H3);
  }
  gru_scan<<<2, 1024, smemScan, stream>>>(xpbuf, whh0B, bhh0,
                                          h, y0cat, hnext);

  // layer 1: input is y0cat (2H), xp buffer reused
  for (int d = 0; d < 2; ++d) {
    gemm_bf16_wmma<<<dim3((H3 + 255) / 256, M_ / 64), 256, 0, stream>>>(
        y0cat, wih1B + (size_t)d * H3 * C2H, bih1 + (size_t)d * H3,
        xpbuf + (size_t)d * M_ * H3, C2H, H3);
  }
  gru_scan<<<2, 1024, smemScan, stream>>>(xpbuf, whh1B, bhh1,
                                          h + 2 * B_ * H_, y1cat,
                                          hnext + 2 * B_ * H_);

  // trailing Linear (dominant GEMM: 16384 x 10000 x 1024) + softmax in place
  gemm_bf16_wmma<<<dim3((V_ + 255) / 256, M_ / 64), 256, 0, stream>>>(
      y1cat, wlinB, blin, out, C2H, V_);
  softmax_rows<<<M_, 256, 0, stream>>>(out);
}
